// SinusoidsGenerator_72868415144384
// MI455X (gfx1250) — matrix operand order repaired and checked by reading, not compiled, not run on hardware
//
#include <hip/hip_runtime.h>
#include <hip/hip_bf16.h>
#include <math.h>

typedef __attribute__((ext_vector_type(16))) _Float16     v16h;
typedef __attribute__((ext_vector_type(8)))  float        v8f;
typedef __attribute__((ext_vector_type(4)))  unsigned int v4u;
typedef __attribute__((ext_vector_type(8)))  int          v8i;
typedef __attribute__((ext_vector_type(4)))  int          v4i;

#define HOP     240
#define FRAMES  500
#define NBATCH  4
#define LSAMP   (FRAMES * HOP)        // 120000
#define NSAMP   (NBATCH * LSAMP)      // 480000
#define NHARM   200
#define FSR     24000.0f
#define TWO_PI_OVER_FS 2.61799387799149436539e-4f  // 2*pi/24000

// ---------------------------------------------------------------------------
// Kernel A: frame-level phase prefix sum.  TDM (tensor_load_to_lds) stages the
// 500-frame f0 contour into LDS as a 512-element tile -- the TDM's OOB-reads-
// return-zero rule supplies the zero padding branchlessly.  Then one
// v_wmma_f32_16x16x32_f16 per 256 frames computes the inclusive prefix of 16
// chunks at once:  D = [T|T](16x32) x [Vhi;Vlo](32x16), T = lower-tri ones,
// with an f16 hi/lo split so the f32 result is exact to ~2^-17.  Cross-chunk
// offsets via a wave32 shuffle scan.  Output: exclusive prefix mod 100
// (240*100 Hz-samples == 2*pi for every integer harmonic).
// ---------------------------------------------------------------------------
__global__ __launch_bounds__(32)
void prefix_wmma_kernel(const float* __restrict__ cf0, float* __restrict__ ebase) {
    __shared__ float ldsf[512];

    const int b      = blockIdx.x;
    const int lane   = threadIdx.x;
    const int col    = lane & 15;
    const bool upper = lane >= 16;

    // ---- TDM: async DMA cf0[b,0:500] -> ldsf[0:500], ldsf[500:512] = 0 ----
    {
        unsigned long long ga = (unsigned long long)(const void*)(cf0 + b * FRAMES);
        unsigned lds_off = (unsigned)(unsigned long long)(const void*)ldsf; // low 32 = LDS offset
        v4u g0 = { 1u,                                  // count=1 (valid user D#)
                   lds_off,                             // [63:32]  lds_addr
                   (unsigned)(ga & 0xffffffffu),        // [95:64]  global_addr lo
                   (unsigned)((ga >> 32) & 0x01ffffffu) // [120:96] global_addr hi
                       | (2u << 30) };                  // [127:126] type = 2 (image)
        v8i g1 = { (int)(2u << 16),                     // data_size = 2 (4 bytes)
                   (int)(500u << 16),                   // tensor_dim0[15:0] (=500)
                   (int)(1u << 16),                     // tensor_dim0 hi, tensor_dim1 = 1
                   (int)(512u << 16),                   // tile_dim0 = 512
                   1,                                   // tile_dim1 = 1
                   500,                                 // tensor_dim0_stride = 500
                   0, 0 };
        v4i gz4 = { 0, 0, 0, 0 };
        v8i gz8 = { 0, 0, 0, 0, 0, 0, 0, 0 };
        __builtin_amdgcn_tensor_load_to_lds(g0, g1, gz4, gz4, gz8, 0);
        __builtin_amdgcn_s_wait_tensorcnt(0);           // s_wait_tensorcnt 0
    }

    // ---- Build A = [T | T] in contiguous K order (B uses the same) ----
    // lane<16 : elems 0..15 -> K=0..15
    // lane>=16: elems 0..15 -> K=16..31
    v16h a;
    const int m = col;                                  // row index M = lane & 15
    #pragma unroll
    for (int e = 0; e < 16; ++e) {
        int k = e + (upper ? 16 : 0);
        int j = k & 15;                                 // T column (A repeats T in K)
        a[e] = (j <= m) ? (_Float16)1.0f : (_Float16)0.0f;
    }

    v8f   res[2];
    float csum[2];
    #pragma unroll
    for (int tile = 0; tile < 2; ++tile) {
        const int base = tile * 256 + col * 16;         // first frame of chunk 'col'
        // B, same contiguous K order: lane n<16 = column n rows K=0..15 (hi parts),
        //                 lane n+16 = column n rows K=16..31 (lo parts).
        v16h bm;
        #pragma unroll
        for (int e = 0; e < 16; ++e) {
            float v = ldsf[base + e];                   // branchless, zero-padded
            _Float16 hi = (_Float16)v;
            _Float16 lo = (_Float16)(v - (float)hi);
            bm[e] = upper ? lo : hi;
        }
        v8f c = {};
        c = __builtin_amdgcn_wmma_f32_16x16x32_f16(
                /*neg_a=*/false, a, /*neg_b=*/false, bm,
                /*c_mod=*/(short)0, c, /*reuse_a=*/false, /*reuse_b=*/false);
        res[tile]  = c;
        csum[tile] = c[7];      // upper lanes: row M=15 = chunk total
    }

    // Chunk totals: chunk n lives in lane (n&15)+16, tile n>>4, element 7.
    float s0 = __shfl(csum[0], col + 16, 32);
    float s1 = __shfl(csum[1], col + 16, 32);
    float cs = upper ? s1 : s0;                         // total of chunk id == lane

    // Inclusive shuffle scan over the 32 chunk totals.
    float incl = cs;
    #pragma unroll
    for (int d = 1; d < 32; d <<= 1) {
        float t = __shfl_up(incl, d, 32);
        if (lane >= d) incl += t;
    }
    float excl = incl - cs;                             // offset for chunk 'lane'

    float off0 = __shfl(excl, col, 32);                 // tile0 chunk col
    float off1 = __shfl(excl, col + 16, 32);            // tile1 chunk col

    // D layout: VGPR r holds rows M=r (lanes 0-15) / M=r+8 (lanes 16-31),
    // column N = lane&15.
    #pragma unroll
    for (int tile = 0; tile < 2; ++tile) {
        const int base = tile * 256 + col * 16;
        const float off = tile ? off1 : off0;
        #pragma unroll
        for (int r = 0; r < 8; ++r) {
            int M = r + (upper ? 8 : 0);
            int f = base + M;
            if (f < FRAMES) {
                float inclusive = res[tile][r] + off;
                float exclusive = inclusive - ldsf[f];  // sum of frames < f
                // exclusive >= 0; reduce mod 100 branchlessly
                float red = fmaf(-100.0f, floorf(exclusive * 0.01f), exclusive);
                ebase[b * FRAMES + f] = red;
            }
        }
    }
}

// ---------------------------------------------------------------------------
// Kernel B: per-sample harmonic synthesis with Chebyshev cosine recurrence.
// phase(l) = (240*E[t] + f0*(k+1)) * 2*pi/24000,  E[t] = sum_{t'<t} cf0 mod 100
// excit = uv * 0.1 * sum_{h=1..hmax} cos(h*phi),  hmax = max h with f0*h<12000
// ---------------------------------------------------------------------------
__global__ __launch_bounds__(256)
void excite_kernel(const float* __restrict__ cf0, const float* __restrict__ uv,
                   const float* __restrict__ ebase, float* __restrict__ out) {
    int i = blockIdx.x * 256 + threadIdx.x;
    if (i >= NSAMP) return;
    int b = i / LSAMP;
    int l = i - b * LSAMP;
    int t = l / HOP;
    int k = l - t * HOP;
    int fi = b * FRAMES + t;

    __builtin_prefetch(&cf0[fi], 0, 3);     // global_prefetch_b8

    float f0 = cf0[fi];
    float u  = uv[fi];
    float val = 0.0f;
    if (u != 0.0f) {
        float s   = fmaf(240.0f, ebase[fi], f0 * (float)(k + 1));   // in [0,120000)
        s         = fmaf(-FSR, floorf(s * (1.0f / FSR)), s);        // mod 24000
        float phi = s * TWO_PI_OVER_FS;

        int hm = (int)(11999.999f / f0);    // strict f0*h < 12000
        if (hm > NHARM) hm = NHARM;

        float cc  = cosf(phi);
        float cp  = 1.0f;
        float tc  = cc + cc;
        float acc = 0.0f;
        for (int h = 1; h <= hm; ++h) {
            acc += cc;
            float nx = fmaf(tc, cc, -cp);   // cos((h+1)phi)
            cp = cc;
            cc = nx;
        }
        val = 0.1f * acc;
    }
    out[i] = val;
}

// ---------------------------------------------------------------------------
// Kernel C: noise = 0.003 * normal(key(1)).  Exact Threefry-2x32 (20 rounds,
// key = (0,1)), counts = iota(480000) split into halves, JAX bits->uniform->
// sqrt(2)*erfinv mapping.
// ---------------------------------------------------------------------------
__device__ __forceinline__ unsigned rotl32(unsigned x, int r) {
    return (x << r) | (x >> (32 - r));
}

__device__ __forceinline__ float bits_to_normal(unsigned bits) {
    unsigned ub = (bits >> 9) | 0x3f800000u;
    float f = __uint_as_float(ub) - 1.0f;          // [0, 1)
    const float lo = -0.99999994f;                 // nextafter(-1, 0)
    float x = f * (1.0f - lo) + lo;                // uniform in [lo, 1)
    return 1.41421356237309504880f * erfinvf(x) * 0.003f;
}

__global__ __launch_bounds__(256)
void noise_kernel(float* __restrict__ out) {
    const int half = NSAMP / 2;                    // 240000
    int j = blockIdx.x * 256 + threadIdx.x;
    if (j >= half) return;

    unsigned x0 = (unsigned)j;
    unsigned x1 = (unsigned)(j + half);
    const unsigned ks0 = 0u, ks1 = 1u;
    const unsigned ks2 = 0x1BD11BDAu ^ ks0 ^ ks1;
    x0 += ks0; x1 += ks1;

#define TF_ROUND(r) { x0 += x1; x1 = rotl32(x1, r); x1 ^= x0; }
    TF_ROUND(13) TF_ROUND(15) TF_ROUND(26) TF_ROUND(6)
    x0 += ks1; x1 += ks2 + 1u;
    TF_ROUND(17) TF_ROUND(29) TF_ROUND(16) TF_ROUND(24)
    x0 += ks2; x1 += ks0 + 2u;
    TF_ROUND(13) TF_ROUND(15) TF_ROUND(26) TF_ROUND(6)
    x0 += ks0; x1 += ks1 + 3u;
    TF_ROUND(17) TF_ROUND(29) TF_ROUND(16) TF_ROUND(24)
    x0 += ks1; x1 += ks2 + 4u;
    TF_ROUND(13) TF_ROUND(15) TF_ROUND(26) TF_ROUND(6)
    x0 += ks2; x1 += ks0 + 5u;
#undef TF_ROUND

    out[NSAMP + j]        = bits_to_normal(x0);
    out[NSAMP + half + j] = bits_to_normal(x1);
}

// ---------------------------------------------------------------------------
extern "C" void kernel_launch(void* const* d_in, const int* in_sizes, int n_in,
                              void* d_out, int out_size, void* d_ws, size_t ws_size,
                              hipStream_t stream) {
    (void)in_sizes; (void)n_in; (void)out_size; (void)ws_size;
    const float* cf0 = (const float*)d_in[0];   // [4,500,1]
    const float* uv  = (const float*)d_in[1];   // [4,500,1]
    float* out   = (float*)d_out;               // excit[480000] ++ noise[480000]
    float* ebase = (float*)d_ws;                // 2000 floats of scratch

    prefix_wmma_kernel<<<NBATCH, 32, 0, stream>>>(cf0, ebase);
    excite_kernel<<<(NSAMP + 255) / 256, 256, 0, stream>>>(cf0, uv, ebase, out);
    noise_kernel<<<(NSAMP / 2 + 255) / 256, 256, 0, stream>>>(out);
}
